// GQ_Attention_57243324121221
// MI455X (gfx1250) — matrix operand order repaired and checked by reading, not compiled, not run on hardware
//
#include <hip/hip_runtime.h>
#include <hip/hip_bf16.h>

typedef _Float16 v16h __attribute__((ext_vector_type(16)));
typedef _Float16 half8 __attribute__((ext_vector_type(8)));
typedef float    v8f  __attribute__((ext_vector_type(8)));

#define N_HEADS 16
#define N_KV    4
#define HD      128
#define TLEN    2048
#define BATCH   2
#define CDIM    2048
#define KVDIM   1024   // 2 * N_KV * HD

// ---- CDNA5 async global->LDS DMA via inline asm (device pass only) ----
#if defined(__gfx1250__)
#define USE_ASYNC_LDS 1
#endif

static __device__ __forceinline__ void wait_async0() {
#if defined(USE_ASYNC_LDS)
    asm volatile("s_wait_asynccnt 0x0" ::: "memory");
#endif
}

// copy 32 contiguous bytes global -> LDS (per lane), async if available
static __device__ __forceinline__ void copy32(const _Float16* g, _Float16* l) {
#if defined(USE_ASYNC_LDS)
    // Flat pointers into the LDS aperture carry the LDS byte offset in
    // addr[31:0] (ISA 10.2), which is exactly what VDST wants here.
    const unsigned lo = (unsigned)(size_t)l;
    asm volatile(
        "global_load_async_to_lds_b128 %0, %1, off\n\t"
        "global_load_async_to_lds_b128 %0, %1, off offset:16"
        :: "v"(lo), "v"((unsigned long long)(size_t)g)
        : "memory");
#else
    half8 t0 = *(const half8*)g;
    half8 t1 = *(const half8*)(g + 8);
    *(half8*)l       = t0;
    *(half8*)(l + 8) = t1;
#endif
}

static __device__ __forceinline__ v16h load16(const _Float16* p0, const _Float16* p1) {
    half8 a = *(const half8*)p0;   // 16B aligned
    half8 b = *(const half8*)p1;   // 16B aligned
    return __builtin_shufflevector(a, b, 0,1,2,3,4,5,6,7,8,9,10,11,12,13,14,15);
}

static __device__ __forceinline__ v8f wmma_f16(v16h a, v16h b, v8f c) {
    return __builtin_amdgcn_wmma_f32_16x16x32_f16(
        /*neg_a=*/false, a, /*neg_b=*/false, b,
        /*c_mod=*/(short)0, c, /*reuse_a=*/false, /*reuse_b=*/false);
}

// ---------------------------------------------------------------------------
// WMMA GEMM:  C[MxN](f32) = A[MxK](f16) * Bt[NxK](f16)^T
// Block 128x128, 256 threads (8 waves), wave tile 64x32 (4x2 wmma tiles).
// Double-buffered K slabs staged with async global->LDS DMA.
// M%128==0, N%128==0, K%32==0 (true for all shapes used here).
// ---------------------------------------------------------------------------
__global__ __launch_bounds__(256) void gemm_wmma(
    const _Float16* __restrict__ A,
    const _Float16* __restrict__ Bt,
    float* __restrict__ C,
    int M, int N, int K)
{
    __shared__ _Float16 As[2][128][40];   // 40-half row stride = 80B (16B aligned)
    __shared__ _Float16 Bs[2][128][40];

    const int tid  = threadIdx.x;
    const int lane = tid & 31;
    const int wave = tid >> 5;
    const int hi   = lane >> 4;    // half-wave select
    const int col  = lane & 15;
    const int rowBase = blockIdx.y * 128;
    const int colBase = blockIdx.x * 128;
    const int wm = (wave >> 2) * 64;   // wave row offset in block
    const int wn = (wave & 3)  * 32;   // wave col offset in block

    v8f acc[4][2];
#pragma unroll
    for (int i = 0; i < 4; ++i)
#pragma unroll
        for (int j = 0; j < 2; ++j) acc[i][j] = (v8f){};

    const int ldRow = tid >> 1;          // 0..127
    const int ldSeg = (tid & 1) * 16;    // 0 or 16 halves

    const _Float16* gA = A  + (size_t)(rowBase + ldRow) * K + ldSeg;
    const _Float16* gB = Bt + (size_t)(colBase + ldRow) * K + ldSeg;

    auto stage = [&](int p, int kb) {
        copy32(gA + kb, &As[p][ldRow][ldSeg]);
        copy32(gB + kb, &Bs[p][ldRow][ldSeg]);
    };

    stage(0, 0);
    int p = 0;
    for (int kb = 0; kb < K; kb += 32) {
        wait_async0();          // my async copies into buffer p have landed
        __syncthreads();        // everyone's copies have landed
        if (kb + 32 < K) stage(p ^ 1, kb + 32);   // overlap next slab DMA

        // ---- fragments from buffer p ----
        v16h a[4], b[2];
#pragma unroll
        for (int mt = 0; mt < 4; ++mt) {
            const _Float16* q = &As[p][wm + mt * 16 + col][hi * 8];
            a[mt] = load16(q, q + 16);   // K = {hi*8..+7} and {16+hi*8..+7}
        }
#pragma unroll
        for (int nt = 0; nt < 2; ++nt) {
            const _Float16* q = &Bs[p][wn + nt * 16 + col][hi * 8];
            b[nt] = load16(q, q + 16);   // K = {hi*8..+7} and {16+hi*8..+7}
        }
#pragma unroll
        for (int mt = 0; mt < 4; ++mt)
#pragma unroll
            for (int nt = 0; nt < 2; ++nt)
                acc[mt][nt] = wmma_f16(a[mt], b[nt], acc[mt][nt]);
        p ^= 1;
    }

    // ---- epilogue: C/D layout = VGPR i -> row i + 8*hi, col = lane&15 ----
#pragma unroll
    for (int mt = 0; mt < 4; ++mt)
#pragma unroll
        for (int nt = 0; nt < 2; ++nt)
#pragma unroll
            for (int i = 0; i < 8; ++i) {
                const int r = rowBase + wm + mt * 16 + i + 8 * hi;
                const int c = colBase + wn + nt * 16 + col;
                C[(size_t)r * N + c] = acc[mt][nt][i];
            }
}

// ---------------------------------------------------------------------------
// Flash attention (causal GQA). Grid: (T/128, H, B). 256 threads = 8 waves,
// wave w owns 16 q rows. Q frags resident; K staged [s][d] via async DMA;
// V staged transposed [d][s]; P converted C-layout -> A-layout via LDS.
// ---------------------------------------------------------------------------
__global__ __launch_bounds__(256) void attn_wmma(
    const _Float16* __restrict__ Qh,   // [B,H,T,D]
    const _Float16* __restrict__ Kh,   // [B,KV,T,D]
    const _Float16* __restrict__ Vh,   // [B,KV,T,D]
    _Float16* __restrict__ Yh)         // [B*T, C], col = h*D + d
{
    __shared__ _Float16 Ks[32][136];      // [s][d], 136 halves = 272B stride
    __shared__ _Float16 Vts[128][40];     // [d][s]
    __shared__ _Float16 Ps[8][16][40];    // per-wave P patch [row][s]

    const int tid  = threadIdx.x;
    const int lane = tid & 31;
    const int wave = tid >> 5;
    const int hi   = lane >> 4;
    const int col  = lane & 15;
    const int qt = blockIdx.x, h = blockIdx.y, b = blockIdx.z;
    const int kvh = h >> 2;                       // G = H/KV = 4
    const int qbase = qt * 128 + wave * 16;       // first q row of this wave

    // Q fragments (kept in registers for whole kernel): 4 k-chunks of 32
    v16h qa[4];
    {
        const _Float16* qrow =
            Qh + (((size_t)(b * N_HEADS + h)) * TLEN + qbase + col) * HD;
#pragma unroll
        for (int kc = 0; kc < 4; ++kc) {
            const _Float16* p = qrow + kc * 32 + hi * 8;
            qa[kc] = load16(p, p + 16);
        }
    }

    float mrow[8], lrow[8];
    v8f o[8];
#pragma unroll
    for (int i = 0; i < 8; ++i) { mrow[i] = -1e30f; lrow[i] = 0.f; o[i] = (v8f){}; }

    const _Float16* kbase = Kh + ((size_t)(b * N_KV + kvh)) * TLEN * HD;
    const _Float16* vbase = Vh + ((size_t)(b * N_KV + kvh)) * TLEN * HD;
    const int send = qt * 128 + 128;
    const int srow = tid >> 3;           // 0..31
    const int sseg = (tid & 7) * 16;     // 0..112

    for (int sblk = 0; sblk < send; sblk += 32) {
        // ---- cooperative staging ----
        {
            copy32(kbase + (size_t)(sblk + srow) * HD + sseg, &Ks[srow][sseg]);
            const _Float16* gv = vbase + (size_t)(sblk + srow) * HD + sseg;
            half8 v0 = *(const half8*)gv;
            half8 v1 = *(const half8*)(gv + 8);
#pragma unroll
            for (int j = 0; j < 8; ++j) {
                Vts[sseg + j][srow]     = v0[j];
                Vts[sseg + 8 + j][srow] = v1[j];
            }
        }
        wait_async0();
        __syncthreads();

        if (sblk <= qbase + 15) {    // skip fully-masked blocks (wave-uniform)
            // ---- S = Q K^T  (16 x 32 per wave) ----
            v8f s0 = (v8f){}, s1 = (v8f){};
#pragma unroll
            for (int kc = 0; kc < 4; ++kc) {
                const _Float16* p0 = &Ks[col][kc * 32 + hi * 8];
                const _Float16* p1 = &Ks[16 + col][kc * 32 + hi * 8];
                v16h b0 = load16(p0, p0 + 16);
                v16h b1 = load16(p1, p1 + 16);
                s0 = wmma_f16(qa[kc], b0, s0);
                s1 = wmma_f16(qa[kc], b1, s1);
            }

            // ---- online softmax over this 32-wide block ----
#pragma unroll
            for (int i = 0; i < 8; ++i) {
                const int r = qbase + i + 8 * hi;
                float x0 = s0[i] * 0.08838834764831845f;  // 1/sqrt(128)
                float x1 = s1[i] * 0.08838834764831845f;
                if (sblk + col > r)      x0 = -1e30f;
                if (sblk + 16 + col > r) x1 = -1e30f;
                float mx = fmaxf(x0, x1);
#pragma unroll
                for (int d = 1; d < 16; d <<= 1)
                    mx = fmaxf(mx, __shfl_xor(mx, d, 32));
                const float mn   = fmaxf(mrow[i], mx);
                const float corr = __expf(mrow[i] - mn);
                const float p0   = __expf(x0 - mn);
                const float p1   = __expf(x1 - mn);
                float rs = p0 + p1;
#pragma unroll
                for (int d = 1; d < 16; d <<= 1)
                    rs += __shfl_xor(rs, d, 32);
                lrow[i] = lrow[i] * corr + rs;
                mrow[i] = mn;
#pragma unroll
                for (int t = 0; t < 8; ++t) o[t][i] *= corr;
                Ps[wave][i + 8 * hi][col]      = (_Float16)p0;
                Ps[wave][i + 8 * hi][16 + col] = (_Float16)p1;
            }
            __builtin_amdgcn_wave_barrier();   // order P stores before reload

            // ---- O += P V  (K=32 over s, N=128 over d) ----
            const _Float16* pp = &Ps[wave][col][hi * 8];
            v16h pa = load16(pp, pp + 16);
#pragma unroll
            for (int dt = 0; dt < 8; ++dt) {
                const _Float16* vp = &Vts[dt * 16 + col][hi * 8];
                v16h vb = load16(vp, vp + 16);
                o[dt] = wmma_f16(pa, vb, o[dt]);
            }
        }
        __syncthreads();
    }

    // ---- normalize and emit f16 in [B*T, C] (ready for output projection) ----
#pragma unroll
    for (int i = 0; i < 8; ++i) {
        const float inv = 1.0f / lrow[i];
        const size_t yrow = (size_t)b * TLEN + qbase + i + 8 * hi;
#pragma unroll
        for (int dt = 0; dt < 8; ++dt)
            Yh[yrow * CDIM + h * HD + dt * 16 + col] =
                (_Float16)(o[dt][i] * inv);
    }
}

// ---------------------------------------------------------------------------
// Elementwise helpers
// ---------------------------------------------------------------------------
__global__ void conv_f32_to_f16(const float* __restrict__ src,
                                _Float16* __restrict__ dst, int n) {
    int i = blockIdx.x * blockDim.x + threadIdx.x;
    if (i < n) dst[i] = (_Float16)src[i];
}

// dst[n*rows + k] = src[k*cols + n]   (rows = K, cols = N)
__global__ void transpose_to_f16(const float* __restrict__ src,
                                 _Float16* __restrict__ dst,
                                 int rows, int cols) {
    int i = blockIdx.x * blockDim.x + threadIdx.x;
    if (i < rows * cols) {
        int k = i / cols, n = i - k * cols;
        dst[(size_t)n * rows + k] = (_Float16)src[i];
    }
}

__global__ void rope_q_kernel(const float* __restrict__ q32,
                              const float* __restrict__ omega,
                              _Float16* __restrict__ qh) {
    int i = blockIdx.x * blockDim.x + threadIdx.x;  // B*T*H*64 = 1<<22
    int d2 = i & 63;
    int h  = (i >> 6) & 15;
    int t  = (i >> 10) & 2047;
    int b  = i >> 21;
    float s, c;
    __sincosf((float)t * omega[d2], &s, &c);
    const float* src = q32 + ((size_t)(b * TLEN + t)) * CDIM + h * HD;
    float x1 = src[d2], x2 = src[d2 + 64];
    _Float16* dst = qh + (((size_t)(b * N_HEADS + h)) * TLEN + t) * HD;
    dst[d2]      = (_Float16)(x1 * c - x2 * s);
    dst[d2 + 64] = (_Float16)(x1 * s + x2 * c);
}

__global__ void rope_k_kernel(const float* __restrict__ kv32,
                              const float* __restrict__ omega,
                              _Float16* __restrict__ kh) {
    int i = blockIdx.x * blockDim.x + threadIdx.x;  // B*T*KV*64 = 1<<20
    int d2 = i & 63;
    int kh_i = (i >> 6) & 3;
    int t  = (i >> 8) & 2047;
    int b  = i >> 19;
    float s, c;
    __sincosf((float)t * omega[d2], &s, &c);
    const float* src = kv32 + ((size_t)(b * TLEN + t)) * KVDIM + kh_i * HD;
    float x1 = src[d2], x2 = src[d2 + 64];
    _Float16* dst = kh + (((size_t)(b * N_KV + kh_i)) * TLEN + t) * HD;
    dst[d2]      = (_Float16)(x1 * c - x2 * s);
    dst[d2 + 64] = (_Float16)(x1 * s + x2 * c);
}

__global__ void pack_v_kernel(const float* __restrict__ kv32,
                              _Float16* __restrict__ vh) {
    int i = blockIdx.x * blockDim.x + threadIdx.x;  // B*T*KV*128 = 1<<21
    int d = i & 127;
    int kh_i = (i >> 7) & 3;
    int t = (i >> 9) & 2047;
    int b = i >> 20;
    vh[(((size_t)(b * N_KV + kh_i)) * TLEN + t) * HD + d] =
        (_Float16)kv32[((size_t)(b * TLEN + t)) * KVDIM + 512 + kh_i * HD + d];
}

// ---------------------------------------------------------------------------
extern "C" void kernel_launch(void* const* d_in, const int* in_sizes, int n_in,
                              void* d_out, int out_size, void* d_ws, size_t ws_size,
                              hipStream_t stream) {
    (void)in_sizes; (void)n_in; (void)out_size; (void)ws_size;
    const float* x     = (const float*)d_in[0];
    const float* wq    = (const float*)d_in[1];
    const float* wkv   = (const float*)d_in[2];
    const float* wproj = (const float*)d_in[3];
    const float* omega = (const float*)d_in[4];

    char* ws = (char*)d_ws;
    size_t off = 0;
    auto alloc = [&](size_t bytes) -> char* {
        char* p = ws + off;
        off += (bytes + 255) & ~(size_t)255;
        return p;
    };
    _Float16* xh   = (_Float16*)alloc((size_t)4096 * 2048 * 2);  // x in f16
    _Float16* wqt  = (_Float16*)alloc((size_t)2048 * 2048 * 2);  // wq^T f16
    _Float16* wkvt = (_Float16*)alloc((size_t)1024 * 2048 * 2);  // wkv^T f16
    _Float16* wpt  = (_Float16*)alloc((size_t)2048 * 2048 * 2);  // wproj^T f16
    float*    q32  = (float*)   alloc((size_t)4096 * 2048 * 4);  // q pre-rope
    float*    kv32 = (float*)   alloc((size_t)4096 * 1024 * 4);  // kv pre-rope
    _Float16* qh   = (_Float16*)alloc((size_t)4096 * 2048 * 2);  // [B,H,T,D]
    _Float16* kh   = (_Float16*)alloc((size_t)BATCH * N_KV * TLEN * HD * 2);
    _Float16* vh   = (_Float16*)alloc((size_t)BATCH * N_KV * TLEN * HD * 2);
    _Float16* yh   = (_Float16*)alloc((size_t)4096 * 2048 * 2);  // attn out f16

    const dim3 blk(256);
    conv_f32_to_f16<<<(4096 * 2048) / 256, blk, 0, stream>>>(x, xh, 4096 * 2048);
    transpose_to_f16<<<(2048 * 2048) / 256, blk, 0, stream>>>(wq,    wqt,  2048, 2048);
    transpose_to_f16<<<(2048 * 1024) / 256, blk, 0, stream>>>(wkv,   wkvt, 2048, 1024);
    transpose_to_f16<<<(2048 * 2048) / 256, blk, 0, stream>>>(wproj, wpt,  2048, 2048);

    gemm_wmma<<<dim3(2048 / 128, 4096 / 128), blk, 0, stream>>>(xh, wqt,  q32,  4096, 2048, 2048);
    gemm_wmma<<<dim3(1024 / 128, 4096 / 128), blk, 0, stream>>>(xh, wkvt, kv32, 4096, 1024, 2048);

    rope_q_kernel<<<(1 << 22) / 256, blk, 0, stream>>>(q32, omega, qh);
    rope_k_kernel<<<(1 << 20) / 256, blk, 0, stream>>>(kv32, omega, kh);
    pack_v_kernel<<<(1 << 21) / 256, blk, 0, stream>>>(kv32, vh);

    attn_wmma<<<dim3(TLEN / 128, N_HEADS, BATCH), blk, 0, stream>>>(qh, kh, vh, yh);

    gemm_wmma<<<dim3(2048 / 128, 4096 / 128), blk, 0, stream>>>(yh, wpt, (float*)d_out,
                                                                4096, 2048, 2048);
}
